// PointNetSetAbstraction_78486232367766
// MI455X (gfx1250) — matrix operand order repaired and checked by reading, not compiled, not run on hardware
//
#include <hip/hip_runtime.h>

// ---------------------------------------------------------------------------
// PointNetSetAbstraction forward for MI455X (gfx1250), wave32 + WMMA bf16 +
// async global->LDS staging.
//
//   K0 wprep_kernel      : W1/W2 f32 -> bf16 in workspace (enables async DMA)
//   K1 fps_kernel        : farthest point sampling, 1 WGP per batch
//   K2 l1_base_kernel    : base[o,b,n]   (layer-1 part independent of s)
//   K3 l1_off_kernel     : off [o,b,s]   (layer-1 part independent of n)
//   K4 bn1_stats_kernel  : analytic BN1 stats -> alpha1/delta1
//   K5 gemm2_kernel      : x1 generated on the fly, WMMA 128x128xK128,
//                          y2 bf16 [p][o] + channel sum/sumsq atomics
//                          (weights staged via global_load_async_to_lds)
//   K6 bn_finalize       : BN2 alpha/delta
//   K6b bn2_apply_kernel : y2 <- ReLU(a2*y2+d2) in place (so gemm3 staging is
//                          a raw async copy, no VALU in the staging path)
//   K7 gemm3_kernel(m=0) : WMMA 256xK128, stats of y3 (async-staged tiles)
//   K8 bn_finalize       : BN3 alpha/delta
//   K9 gemm3_kernel(m=1) : fused BN3+ReLU+max-over-n -> feat output
// ---------------------------------------------------------------------------

#define BATCH   8
#define NPTS    16384
#define SPTS    2048        // NPOINT
#define NSAMP   32
#define PTOT    (BATCH*NSAMP*SPTS)   // 524288 "pixels" through the MLP
#define BN_EPS  1e-5f

// workspace byte offsets (total ~143 MB, fits in L2)
#define NX_OFF     0                      // new_xyz  [B][S][3] f32    196608 B
#define BASE1_OFF  196608                 // base1 [128][B][32] f32    131072 B
#define OFF1_OFF   327680                 // off1  [128][B][S]  f32   8388608 B
#define ST1_OFF    8716288                // alpha1[128], delta1[128]    1024 B
#define ST2_OFF    8717312                // sum,sq,alpha,delta (128ea)  2048 B
#define ST3_OFF    8719360                // sum,sq,alpha,delta (256ea)  4096 B
#define W1BF_OFF   8723456                // W1 bf16 [128][128]         32768 B
#define W2BF_OFF   8756224                // W2 bf16 [256][128]         65536 B
#define Y2_OFF     8821760                // y2/x2 bf16 [P][128]    134217728 B

typedef __attribute__((ext_vector_type(8)))  float  v8f;
typedef __attribute__((ext_vector_type(8)))  __bf16 v8bf;
typedef __attribute__((ext_vector_type(16))) __bf16 v16bf;

union U128 { uint4 v; unsigned short h[8]; };

__device__ __forceinline__ unsigned short bfbits(float f) {
    unsigned u = __builtin_bit_cast(unsigned, f);
    unsigned r = (u + 0x7FFFu + ((u >> 16) & 1u)) >> 16;   // RNE
    return (unsigned short)r;
}
__device__ __forceinline__ float bf2f(unsigned short h) {
    unsigned u = ((unsigned)h) << 16;
    return __builtin_bit_cast(float, u);
}
// load a 16-element bf16 fragment as two contiguous 16B LDS reads
__device__ __forceinline__ v16bf ld_frag(const unsigned short* p, int off2) {
    v8bf lo = *(const v8bf*)p;
    v8bf hi = *(const v8bf*)(p + off2);
    return __builtin_shufflevector(lo, hi, 0,1,2,3,4,5,6,7,8,9,10,11,12,13,14,15);
}
__device__ __forceinline__ float blockReduceSum(float v, float* red) {
    int t = threadIdx.x;
    red[t] = v; __syncthreads();
    for (int k = blockDim.x >> 1; k > 0; k >>= 1) {
        if (t < k) red[t] += red[t + k];
        __syncthreads();
    }
    float r = red[0]; __syncthreads();
    return r;
}

// ---------------------------------------------------------------------------
// Async global->LDS 16B copy (CDNA5: GLOBAL_LOAD_ASYNC_TO_LDS_B128, ASYNCcnt).
// Builtin signature (from hipcc diagnostics): arg0 = native clang vector
// 'int __attribute__((vector_size(16)))' pointer in addrspace(1) (global),
// arg1 = same pointee in addrspace(3) (LDS), then imm offset and imm cpol.
// LDS operand = low 32 bits of the generic address (ISA 10.2).
// ---------------------------------------------------------------------------
typedef int v4i_native __attribute__((vector_size(16)));
typedef __attribute__((address_space(1))) v4i_native as1_v4i;
typedef __attribute__((address_space(3))) v4i_native as3_v4i;

__device__ __forceinline__ void async_copy16(const void* g, void* l) {
#if __has_builtin(__builtin_amdgcn_global_load_async_to_lds_b128)
    __builtin_amdgcn_global_load_async_to_lds_b128(
        (as1_v4i*)(unsigned long long)(uintptr_t)g,
        (as3_v4i*)(unsigned)(uintptr_t)l, 0, 0);
#else
    asm volatile("global_load_async_to_lds_b128 %0, %1, off"
                 :: "v"((unsigned)(uintptr_t)l),
                    "v"((unsigned long long)(uintptr_t)g)
                 : "memory");
#endif
}
__device__ __forceinline__ void async_wait0() {
#if __has_builtin(__builtin_amdgcn_s_wait_asynccnt)
    __builtin_amdgcn_s_wait_asynccnt(0);
#else
    asm volatile("s_wait_asynccnt 0" ::: "memory");
#endif
}

// ---------------------------------------------------------------------------
// K0: weights f32 -> bf16 (once, so GEMM staging is a raw async byte copy)
// ---------------------------------------------------------------------------
__global__ void wprep_kernel(const float* __restrict__ W1, const float* __restrict__ W2,
                             unsigned short* __restrict__ w1bf,
                             unsigned short* __restrict__ w2bf) {
    int t = blockIdx.x * 256 + threadIdx.x;
    if (t < 128 * 128) w1bf[t] = bfbits(W1[t]);
    if (t < 256 * 128) w2bf[t] = bfbits(W2[t]);
}

// ---------------------------------------------------------------------------
// K1: FPS. One 1024-thread workgroup (32 waves) per batch; 16384 points live
// in registers (16/thread); per step: dist update + wave32 shfl argmax +
// cross-wave LDS argmax; register owner of the winner broadcasts via LDS.
// ---------------------------------------------------------------------------
__global__ __launch_bounds__(1024)
void fps_kernel(const float* __restrict__ xyz, float* __restrict__ out_newxyz,
                float* __restrict__ nxws) {
    const int b = blockIdx.x;
    const int t = threadIdx.x;
    const int lane = t & 31, wv = t >> 5;

    __shared__ float sc[3];
    __shared__ int   sfar;
    __shared__ float wbest[32];
    __shared__ int   widx[32];

    const float* xb = xyz + (size_t)b * 3 * NPTS;
    float px[16], py[16], pz[16], dmin[16];
#pragma unroll
    for (int i = 0; i < 16; ++i) {
        int p = t + 1024 * i;
        px[i] = xb[p]; py[i] = xb[NPTS + p]; pz[i] = xb[2 * NPTS + p];
        dmin[i] = 1e10f;
    }
    if (t == 0) {
        int f0 = (int)((2654435761u * (unsigned)(b + 1)) % (unsigned)NPTS);
        sfar = f0;
        sc[0] = xb[f0]; sc[1] = xb[NPTS + f0]; sc[2] = xb[2 * NPTS + f0];
    }
    __syncthreads();

    for (int j = 0; j < SPTS; ++j) {
        float cx = sc[0], cy = sc[1], cz = sc[2];
        if (t == 0) {   // record centroid j
            out_newxyz[b * 3 * SPTS + j]            = cx;
            out_newxyz[b * 3 * SPTS + SPTS + j]     = cy;
            out_newxyz[b * 3 * SPTS + 2 * SPTS + j] = cz;
            float* nx = nxws + ((size_t)b * SPTS + j) * 3;
            nx[0] = cx; nx[1] = cy; nx[2] = cz;
        }
        float best = -1.0f; int bi = 0;
#pragma unroll
        for (int i = 0; i < 16; ++i) {
            float dx = px[i] - cx, dy = py[i] - cy, dz = pz[i] - cz;
            float d = dx * dx + dy * dy + dz * dz;
            dmin[i] = fminf(dmin[i], d);
            if (dmin[i] > best) { best = dmin[i]; bi = t + 1024 * i; }
        }
#pragma unroll
        for (int m = 16; m > 0; m >>= 1) {
            float ob = __shfl_xor(best, m, 32);
            int   oi = __shfl_xor(bi,   m, 32);
            if (ob > best) { best = ob; bi = oi; }
        }
        if (lane == 0) { wbest[wv] = best; widx[wv] = bi; }
        __syncthreads();
        if (t < 32) {
            float bb = wbest[t]; int ii = widx[t];
#pragma unroll
            for (int m = 16; m > 0; m >>= 1) {
                float ob = __shfl_xor(bb, m, 32);
                int   oi = __shfl_xor(ii, m, 32);
                if (ob > bb) { bb = ob; ii = oi; }
            }
            if (t == 0) sfar = ii;
        }
        __syncthreads();
        {
            int nf = sfar;
            if ((nf & 1023) == t) {
                int i = nf >> 10;
                sc[0] = px[i]; sc[1] = py[i]; sc[2] = pz[i];
            }
        }
        __syncthreads();
    }
}

// K2: base1[o,b,n] = b0[o] + sum_c W0[o,c]*in[c,b,n]   (first 32 points)
__global__ void l1_base_kernel(const float* __restrict__ xyz,
                               const float* __restrict__ pts,
                               const float* __restrict__ W0,
                               const float* __restrict__ b0,
                               float* __restrict__ base1) {
    const int o = blockIdx.x, b = blockIdx.y, n = threadIdx.x;   // n < 32
    const float* w = W0 + o * 67;
    float v = b0[o];
#pragma unroll
    for (int c = 0; c < 3; ++c)
        v = fmaf(w[c], xyz[(size_t)b * 3 * NPTS + c * NPTS + n], v);
    for (int c = 0; c < 64; ++c)
        v = fmaf(w[3 + c], pts[(size_t)b * 64 * NPTS + c * NPTS + n], v);
    base1[(o * BATCH + b) * NSAMP + n] = v;
}

// K3: off1[o,b,s] = -(W0[o,0..2] . new_xyz[b,s])
__global__ void l1_off_kernel(const float* __restrict__ W0,
                              const float* __restrict__ nxws,
                              float* __restrict__ off1) {
    const int o = blockIdx.x, b = blockIdx.y;
    const float w0 = W0[o * 67], w1 = W0[o * 67 + 1], w2 = W0[o * 67 + 2];
    for (int s = threadIdx.x; s < SPTS; s += blockDim.x) {
        const float* nx = nxws + ((size_t)b * SPTS + s) * 3;
        off1[(o * BATCH + b) * SPTS + s] = -(w0 * nx[0] + w1 * nx[1] + w2 * nx[2]);
    }
}

// K4: analytic BN1 stats (y1 = base(b,n)+off(b,s) separates over n,s)
__global__ void bn1_stats_kernel(const float* __restrict__ base1,
                                 const float* __restrict__ off1,
                                 const float* __restrict__ g,
                                 const float* __restrict__ be,
                                 float* __restrict__ st1) {
    const int o = blockIdx.x, t = threadIdx.x;
    __shared__ float red[256];
    float mu_acc = 0.f, e2_acc = 0.f;
    for (int b = 0; b < BATCH; ++b) {
        const float* orow = off1 + ((size_t)o * BATCH + b) * SPTS;
        float s = 0.f, q = 0.f;
        for (int i = t; i < SPTS; i += 256) { float v = orow[i]; s += v; q += v * v; }
        float so = blockReduceSum(s, red);
        float qo = blockReduceSum(q, red);
        float sb = 0.f, qb = 0.f;
        if (t < NSAMP) { float v = base1[(o * BATCH + b) * NSAMP + t]; sb = v; qb = v * v; }
        float sB = blockReduceSum(sb, red);
        float qB = blockReduceSum(qb, red);
        float mb = sB * (1.f / NSAMP), mo = so * (1.f / SPTS);
        mu_acc += mb + mo;
        e2_acc += qB * (1.f / NSAMP) + 2.f * mb * mo + qo * (1.f / SPTS);
    }
    if (t == 0) {
        float mu = mu_acc * (1.f / BATCH);
        float var = e2_acc * (1.f / BATCH) - mu * mu;
        float a = g[o] * rsqrtf(var + BN_EPS);
        st1[o] = a; st1[128 + o] = be[o] - a * mu;
    }
}

// K6/K8: finalize BN from accumulated sum/sumsq
__global__ void bn_finalize_kernel(float* __restrict__ st, const float* __restrict__ g,
                                   const float* __restrict__ be, int C, float inv_np) {
    int o = threadIdx.x;
    if (o < C) {
        float mean = st[o] * inv_np;
        float var  = st[C + o] * inv_np - mean * mean;
        float a = g[o] * rsqrtf(var + BN_EPS);
        st[2 * C + o] = a;
        st[3 * C + o] = be[o] - a * mean;
    }
}

// K6b: y2 <- ReLU(alpha2*y2 + delta2) in place (vectorized b128)
__global__ __launch_bounds__(256)
void bn2_apply_kernel(unsigned short* __restrict__ y2, const float* __restrict__ st2) {
    __shared__ float sa[128], sd[128];
    const int t = threadIdx.x;
    if (t < 128) { sa[t] = st2[256 + t]; sd[t] = st2[384 + t]; }
    __syncthreads();
    const size_t nch = (size_t)PTOT * 128 / 8;   // 8388608 chunks of 8 elements
    for (size_t c = (size_t)blockIdx.x * 256 + t; c < nch; c += (size_t)gridDim.x * 256) {
        const int k0 = (int)((c * 8) & 127);
        U128 u; u.v = *(uint4*)(y2 + c * 8);
#pragma unroll
        for (int i = 0; i < 8; ++i) {
            float x = fmaf(sa[k0 + i], bf2f(u.h[i]), sd[k0 + i]);
            u.h[i] = bfbits(fmaxf(x, 0.f));
        }
        *(uint4*)(y2 + c * 8) = u.v;
    }
}

// ---------------------------------------------------------------------------
// K5: GEMM2.  y2[o,p] = W1[o,k]*x1[k,p] + b1[o];  x1 built in LDS on the fly.
// W1 tile staged with async global->LDS DMA (raw bf16 copy).
// ---------------------------------------------------------------------------
__global__ __launch_bounds__(256)
void gemm2_kernel(const unsigned short* __restrict__ w1bf, const float* __restrict__ b1,
                  const float* __restrict__ base1, const float* __restrict__ off1,
                  const float* __restrict__ st1,
                  unsigned short* __restrict__ y2, float* __restrict__ st2) {
    extern __shared__ __align__(16) unsigned char smem[];
    unsigned short* sW  = (unsigned short*)smem;                  // [128][128]
    unsigned short* sX  = sW + 128 * 128;                         // [128pt][128k]
    float*          sA1 = (float*)(sX + 128 * 128);               // alpha1[128]
    float*          sBB = sA1 + 128;                              // a*base+delta [128]

    const int t = threadIdx.x;
    const int pbase = blockIdx.x * 128;
    const int b = pbase >> 16, n = (pbase >> 11) & 31, sbase = pbase & 2047;

    // async-stage W1 (32 KB linear copy, 8 x b128 per thread)
#pragma unroll
    for (int i = 0; i < 8; ++i) {
        const int L = (t + i * 256) * 16;       // byte offset
        async_copy16((const char*)w1bf + L, (char*)sW + L);
    }
    if (t < 128) {
        float a = st1[t];
        sA1[t] = a;
        sBB[t] = fmaf(a, base1[(t * BATCH + b) * NSAMP + n], st1[128 + t]);
    }
    __syncthreads();
    for (int idx = t; idx < 128 * 128; idx += 256) {      // x1 tile, ReLU fused
        int k = idx >> 7, sl = idx & 127;
        float x = fmaf(sA1[k], off1[((size_t)k * BATCH + b) * SPTS + sbase + sl], sBB[k]);
        sX[sl * 128 + k] = bfbits(fmaxf(x, 0.f));
    }
    async_wait0();
    __syncthreads();

    const int lane = t & 31, w = t >> 5;
    const int arow  = w * 16 + (lane & 15);
    const int akoff = (lane >> 4) * 8;     // A: lanes 0-15 K{0-7,16-23}, 16-31 K{8-15,24-31}
    const int bkoff = (lane >> 4) * 8;     // B: lanes 0-15 K{0-7,16-23}, 16-31 K{8-15,24-31}
    const int obase = w * 16 + 8 * (lane >> 4);

    v16bf af[4];
#pragma unroll
    for (int kk = 0; kk < 4; ++kk)
        af[kk] = ld_frag(sW + arow * 128 + kk * 32 + akoff, 16);

    float biasv[8];
#pragma unroll
    for (int r = 0; r < 8; ++r) biasv[r] = b1[obase + r];

    float sumr[8] = {0,0,0,0,0,0,0,0}, sqr[8] = {0,0,0,0,0,0,0,0};
#pragma unroll
    for (int ns = 0; ns < 8; ++ns) {
        const int ptl = ns * 16 + (lane & 15);
        v8f acc = {0.f,0.f,0.f,0.f,0.f,0.f,0.f,0.f};
#pragma unroll
        for (int kk = 0; kk < 4; ++kk) {
            v16bf bf = ld_frag(sX + ptl * 128 + kk * 32 + bkoff, 16);
            acc = __builtin_amdgcn_wmma_f32_16x16x32_bf16(
                      false, af[kk], false, bf, (short)0, acc, false, false);
        }
        U128 pk;
#pragma unroll
        for (int r = 0; r < 8; ++r) {
            float v = acc[r] + biasv[r];
            pk.h[r] = bfbits(v);
            sumr[r] += v; sqr[r] += v * v;
        }
        *(uint4*)(y2 + (size_t)(pbase + ptl) * 128 + obase) = pk.v;
    }
#pragma unroll
    for (int r = 0; r < 8; ++r) {          // reduce over the 16 s-lanes, then atomics
        float s = sumr[r], q = sqr[r];
#pragma unroll
        for (int m = 1; m < 16; m <<= 1) {
            s += __shfl_xor(s, m, 32); q += __shfl_xor(q, m, 32);
        }
        if ((lane & 15) == 0) {
            atomicAdd(&st2[obase + r], s);
            atomicAdd(&st2[128 + obase + r], q);
        }
    }
}

// ---------------------------------------------------------------------------
// K7/K9: GEMM3.  y3[o,p] = W2[o,k]*x2[k,p] + b2[o].  x2 (= BN2+ReLU-applied
// y2) and W2 are staged entirely with async global->LDS DMA — zero VALU in
// the staging path, VALU free to co-execute with WMMA.
// Tile: one (b, 16-s slab) x all 32 n = 512 points, M=256.
// mode 0: BN3 sum/sumsq atomics.  mode 1: BN3+ReLU+max-over-n -> feat.
// ---------------------------------------------------------------------------
__global__ __launch_bounds__(256)
void gemm3_kernel(const unsigned short* __restrict__ w2bf, const float* __restrict__ b2,
                  const unsigned short* __restrict__ x2,
                  float* __restrict__ st3, float* __restrict__ feat, int mode) {
    extern __shared__ __align__(16) unsigned char smem[];
    unsigned short* sA = (unsigned short*)smem;                   // W2 [256][128]
    unsigned short* sX = sA + 256 * 128;                          // x2 [512pt][128k]

    const int t = threadIdx.x;
    const int s0 = blockIdx.x * 16, b = blockIdx.y;

    // async-stage W2 (64 KB linear, 16 x b128 per thread)
#pragma unroll
    for (int i = 0; i < 16; ++i) {
        const int L = (t + i * 256) * 16;
        async_copy16((const char*)w2bf + L, (char*)sA + L);
    }
    // async-stage x2 tile (128 KB = 32 chunks of 4 KB, one chunk per n)
    const size_t tilebase = ((size_t)b * 65536 + s0) * 128;   // elements
#pragma unroll
    for (int i = 0; i < 32; ++i) {
        const int L = (t + i * 256) * 16;       // byte offset in tile
        const int nchunk = L >> 12;             // 4 KB per n (16 rows x 256 B)
        const int within = L & 4095;
        const unsigned short* g = x2 + tilebase + (size_t)nchunk * 2048 * 128 + within / 2;
        async_copy16(g, (char*)sX + L);
    }
    async_wait0();
    __syncthreads();

    const int lane = t & 31, w = t >> 5;
    const int akoff = (lane >> 4) * 8;
    const int bkoff = (lane >> 4) * 8;

    v16bf af[2][4];
    int   ob[2];
    float b2v[2][8], a3v[2][8], d3v[2][8];
#pragma unroll
    for (int mc = 0; mc < 2; ++mc) {
        const int row = w * 32 + mc * 16 + (lane & 15);
#pragma unroll
        for (int kk = 0; kk < 4; ++kk)
            af[mc][kk] = ld_frag(sA + row * 128 + kk * 32 + akoff, 16);
        ob[mc] = w * 32 + mc * 16 + 8 * (lane >> 4);
#pragma unroll
        for (int r = 0; r < 8; ++r) {
            b2v[mc][r] = b2[ob[mc] + r];
            if (mode) { a3v[mc][r] = st3[512 + ob[mc] + r]; d3v[mc][r] = st3[768 + ob[mc] + r]; }
        }
    }

    float runmax[2][8] = {{0}}, sumr[2][8] = {{0}}, sqr[2][8] = {{0}};
    for (int nn = 0; nn < 32; ++nn) {
        const int ptl = nn * 16 + (lane & 15);
        v16bf bf[4];
#pragma unroll
        for (int kk = 0; kk < 4; ++kk)
            bf[kk] = ld_frag(sX + ptl * 128 + kk * 32 + bkoff, 16);
#pragma unroll
        for (int mc = 0; mc < 2; ++mc) {
            v8f acc = {0.f,0.f,0.f,0.f,0.f,0.f,0.f,0.f};
#pragma unroll
            for (int kk = 0; kk < 4; ++kk)
                acc = __builtin_amdgcn_wmma_f32_16x16x32_bf16(
                          false, af[mc][kk], false, bf[kk], (short)0, acc, false, false);
#pragma unroll
            for (int r = 0; r < 8; ++r) {
                float v = acc[r] + b2v[mc][r];
                if (mode == 0) { sumr[mc][r] += v; sqr[mc][r] += v * v; }
                else {
                    float x = fmaxf(fmaf(a3v[mc][r], v, d3v[mc][r]), 0.f);
                    runmax[mc][r] = fmaxf(runmax[mc][r], x);   // max over n
                }
            }
        }
    }
    if (mode == 0) {
#pragma unroll
        for (int mc = 0; mc < 2; ++mc)
#pragma unroll
            for (int r = 0; r < 8; ++r) {
                float s = sumr[mc][r], q = sqr[mc][r];
#pragma unroll
                for (int m = 1; m < 16; m <<= 1) {
                    s += __shfl_xor(s, m, 32); q += __shfl_xor(q, m, 32);
                }
                if ((lane & 15) == 0) {
                    atomicAdd(&st3[ob[mc] + r], s);
                    atomicAdd(&st3[256 + ob[mc] + r], q);
                }
            }
    } else {
        const int s = s0 + (lane & 15);
#pragma unroll
        for (int mc = 0; mc < 2; ++mc)
#pragma unroll
            for (int r = 0; r < 8; ++r)
                feat[(size_t)b * 256 * SPTS + (size_t)(ob[mc] + r) * SPTS + s] = runmax[mc][r];
    }
}

// ---------------------------------------------------------------------------
extern "C" void kernel_launch(void* const* d_in, const int* in_sizes, int n_in,
                              void* d_out, int out_size, void* d_ws, size_t ws_size,
                              hipStream_t stream) {
    (void)in_sizes; (void)n_in; (void)out_size; (void)ws_size;
    const float* xyz = (const float*)d_in[0];
    const float* pts = (const float*)d_in[1];
    const float* W0  = (const float*)d_in[2];
    const float* b0  = (const float*)d_in[3];
    const float* g0  = (const float*)d_in[4];
    const float* be0 = (const float*)d_in[5];
    const float* W1  = (const float*)d_in[6];
    const float* b1  = (const float*)d_in[7];
    const float* g1  = (const float*)d_in[8];
    const float* be1 = (const float*)d_in[9];
    const float* W2  = (const float*)d_in[10];
    const float* b2  = (const float*)d_in[11];
    const float* g2  = (const float*)d_in[12];
    const float* be2 = (const float*)d_in[13];

    float* out = (float*)d_out;                       // [0,49152): new_xyz, rest: feat
    char*  ws  = (char*)d_ws;
    float*          nxws  = (float*)(ws + NX_OFF);
    float*          base1 = (float*)(ws + BASE1_OFF);
    float*          off1  = (float*)(ws + OFF1_OFF);
    float*          st1   = (float*)(ws + ST1_OFF);
    float*          st2   = (float*)(ws + ST2_OFF);
    float*          st3   = (float*)(ws + ST3_OFF);
    unsigned short* w1bf  = (unsigned short*)(ws + W1BF_OFF);
    unsigned short* w2bf  = (unsigned short*)(ws + W2BF_OFF);
    unsigned short* y2    = (unsigned short*)(ws + Y2_OFF);

    const size_t SMEM2 = (size_t)128 * 128 * 2 * 2 + 256 * 4;                 //  66.5 KB
    const size_t SMEM3 = (size_t)256 * 128 * 2 + (size_t)512 * 128 * 2;       //   192 KB (WGP: 320 KB)
    (void)hipFuncSetAttribute((const void*)gemm2_kernel,
                              hipFuncAttributeMaxDynamicSharedMemorySize, (int)SMEM2);
    (void)hipFuncSetAttribute((const void*)gemm3_kernel,
                              hipFuncAttributeMaxDynamicSharedMemorySize, (int)SMEM3);

    // zero the BN accumulators (st2 + st3 are contiguous)
    (void)hipMemsetAsync(st2, 0, (512 + 1024) * sizeof(float), stream);

    wprep_kernel<<<128, 256, 0, stream>>>(W1, W2, w1bf, w2bf);
    fps_kernel<<<BATCH, 1024, 0, stream>>>(xyz, out, nxws);
    l1_base_kernel<<<dim3(128, BATCH), NSAMP, 0, stream>>>(xyz, pts, W0, b0, base1);
    l1_off_kernel<<<dim3(128, BATCH), 256, 0, stream>>>(W0, nxws, off1);
    bn1_stats_kernel<<<128, 256, 0, stream>>>(base1, off1, g0, be0, st1);
    gemm2_kernel<<<PTOT / 128, 256, SMEM2, stream>>>(w1bf, b1, base1, off1, st1, y2, st2);
    bn_finalize_kernel<<<1, 128, 0, stream>>>(st2, g1, be1, 128, 1.f / (float)PTOT);
    bn2_apply_kernel<<<8192, 256, 0, stream>>>(y2, st2);
    gemm3_kernel<<<dim3(SPTS / 16, BATCH), 256, SMEM3, stream>>>(
        w2bf, b2, y2, st3, out + 49152, 0);
    bn_finalize_kernel<<<1, 256, 0, stream>>>(st3, g2, be2, 256, 1.f / (float)PTOT);
    gemm3_kernel<<<dim3(SPTS / 16, BATCH), 256, SMEM3, stream>>>(
        w2bf, b2, y2, st3, out + 49152, 1);
}
